// Model_88459146428516
// MI455X (gfx1250) — matrix operand order repaired and checked by reading, not compiled, not run on hardware
//
#include <hip/hip_runtime.h>
#include <hip/hip_bf16.h>
#include <stdint.h>

typedef __attribute__((ext_vector_type(16))) _Float16 v16h;
typedef __attribute__((ext_vector_type(8)))  _Float16 v8h;
typedef __attribute__((ext_vector_type(8)))  float    v8f;

#define DIMP 96      // 68 padded to multiple of 32
#define NHF  64
#define KNBR 32

// ---------------------------------------------------------------------------
// WMMA fragment helpers (gfx1250 V_WMMA_F32_16X16X32_F16 layouts, ISA 7.12.2)
// A (16x32 f16): lane m (0-15)/m+16 holds row m; halves h -> K = hi*16 + h (this code's order; ISA table differs).
// B (32x16 f16): lane n/n+16 holds column n; halves h -> K = hi*16 + h (this code's order; ISA table differs).
// D (16x16 f32): VGPR r: lanes 0-15 -> M=r, lanes 16-31 -> M=8+r; N = lane&15.
// ---------------------------------------------------------------------------
__device__ __forceinline__ int a_k_of_h(int hi, int h) {
  return hi * 16 + h;
}

// rowbase points at X[row, kstep*32]; row-major f16, 16B-aligned chunks.
__device__ __forceinline__ v16h load_a_frag(const _Float16* rowbase, int hi) {
  v8h lo = *(const v8h*)(rowbase + hi * 16);
  v8h hi8 = *(const v8h*)(rowbase + hi * 16 + 8);
  v16h r;
#pragma unroll
  for (int h = 0; h < 8; ++h) { r[h] = lo[h]; r[h + 8] = hi8[h]; }
  return r;
}

__device__ __forceinline__ v8f wmma_f16(v16h a, v16h b, v8f c) {
  return __builtin_amdgcn_wmma_f32_16x16x32_f16(false, a, false, b, (short)0, c,
                                                false, false);
}

// ---------------------------------------------------------------------------
// Weight prep: W [K,N] f32 row-major  ->  WT [Np,Kp] f16 (transposed, padded)
// ---------------------------------------------------------------------------
__global__ void prep_wt_kernel(const float* __restrict__ W, _Float16* __restrict__ WT,
                               int K, int N, int Kp, int Np) {
  int i = blockIdx.x * blockDim.x + threadIdx.x;
  if (i >= Kp * Np) return;
  int k = i % Kp, n = i / Kp;
  float v = (k < K && n < N) ? W[k * N + n] : 0.0f;
  WT[i] = (_Float16)v;
}

// f32 [M,K] -> f16 [M,Kp] zero-padded
__global__ void pack_kernel(const float* __restrict__ X, _Float16* __restrict__ Y,
                            int M, int K, int Kp) {
  size_t i = (size_t)blockIdx.x * blockDim.x + threadIdx.x;
  if (i >= (size_t)M * Kp) return;
  int k = (int)(i % Kp);
  size_t m = i / Kp;
  Y[i] = (_Float16)((k < K) ? X[m * K + k] : 0.0f);
}

// ---------------------------------------------------------------------------
// Generic WMMA GEMM: Y = act(X @ W + b). One wave per 16-row tile.
// Shape-templated: NK = Kp/32 (1..4), NCT = Np/16. Compile-time trip counts
// give straight-line global_load_b128 -> v_wmma sequences (no runtime guards).
//   A   : f16 [M, NK*32]
//   WT  : f16 [NCT*16, NK*32]   (transposed weights)
//   outH: optional f16 [M, NpOut] (NpOut == Np), outF: optional f32 [M, Nreal]
// ---------------------------------------------------------------------------
template <int NK, int NCT>
__global__ __launch_bounds__(256)
void gemm_t_kernel(const _Float16* __restrict__ A, const _Float16* __restrict__ WT,
                   const float* __restrict__ bias, int M, int Nreal,
                   _Float16* __restrict__ outH, int NpOut,
                   float* __restrict__ outF, int doRelu) {
  constexpr int Kp = NK * 32;
  int wave = threadIdx.x >> 5;
  int lane = threadIdx.x & 31;
  int tile = blockIdx.x * (blockDim.x >> 5) + wave;
  if (tile >= (M >> 4)) return;
  int row0 = tile << 4;
  int lr = lane & 15;
  int hi = lane >> 4;
  const _Float16* arow = A + (size_t)(row0 + lr) * Kp;
  v16h afr[NK];
#pragma unroll
  for (int k = 0; k < NK; ++k) afr[k] = load_a_frag(arow + k * 32, hi);
#pragma unroll
  for (int c = 0; c < NCT; ++c) {
    v8f acc = {};
    const _Float16* wrow = WT + (size_t)(c * 16 + lr) * Kp;
#pragma unroll
    for (int k = 0; k < NK; ++k) {
      v16h b = *(const v16h*)(wrow + k * 32 + hi * 16);
      acc = wmma_f16(afr[k], b, acc);
    }
    int n = c * 16 + lr;
    float bv = (n < Nreal) ? bias[n] : 0.0f;
#pragma unroll
    for (int r = 0; r < 8; ++r) {
      int m = row0 + r + hi * 8;
      float v = acc[r] + bv;
      if (doRelu) v = fmaxf(v, 0.0f);
      if (outH) outH[(size_t)m * NpOut + n] = (_Float16)v;
      if (outF && n < Nreal) outF[(size_t)m * Nreal + n] = v;
    }
  }
}

// ---------------------------------------------------------------------------
// Farthest point sampling (sequential argmax chain) - single workgroup.
// Tie-break: smallest index (matches jnp.argmax).
// ---------------------------------------------------------------------------
#define FPS_T 1024
__global__ __launch_bounds__(FPS_T)
void fps_kernel(const float* __restrict__ pos, int n, int ns,
                int* __restrict__ outIdx, float* __restrict__ mind) {
  __shared__ float sv[FPS_T];
  __shared__ int si[FPS_T];
  __shared__ int slast;
  int tid = threadIdx.x;
  if (tid == 0) { outIdx[0] = 0; slast = 0; }
  for (int j = tid; j < n; j += FPS_T) mind[j] = 3.402823e38f;
  __syncthreads();
  for (int s = 1; s < ns; ++s) {
    int last = slast;
    float px = pos[last * 3 + 0], py = pos[last * 3 + 1], pz = pos[last * 3 + 2];
    float best = -1.0f;
    int bidx = 0x7fffffff;
    for (int j = tid; j < n; j += FPS_T) {
      float dx = pos[j * 3 + 0] - px;
      float dy = pos[j * 3 + 1] - py;
      float dz = pos[j * 3 + 2] - pz;
      float d = dx * dx + dy * dy + dz * dz;
      float m = fminf(mind[j], d);
      mind[j] = m;
      if (m > best) { best = m; bidx = j; }
    }
    sv[tid] = best;
    si[tid] = bidx;
    __syncthreads();
    for (int o = FPS_T / 2; o > 0; o >>= 1) {
      if (tid < o) {
        float ov = sv[tid + o];
        int oi = si[tid + o];
        if (ov > sv[tid] || (ov == sv[tid] && oi < si[tid])) { sv[tid] = ov; si[tid] = oi; }
      }
      __syncthreads();
    }
    if (tid == 0) { slast = si[0]; outIdx[s] = si[0]; }
    __syncthreads();
  }
}

__global__ void gather_kernel(const float* __restrict__ pos, const float* __restrict__ nrm,
                              const int* __restrict__ idx, int ns,
                              float* __restrict__ qpos, float* __restrict__ qnrm) {
  int i = blockIdx.x * blockDim.x + threadIdx.x;
  if (i >= ns) return;
  int j = idx[i];
#pragma unroll
  for (int c = 0; c < 3; ++c) {
    qpos[i * 3 + c] = pos[j * 3 + c];
    qnrm[i * 3 + c] = nrm[j * 3 + c];
  }
}

// ---------------------------------------------------------------------------
// Ball query: per query, 32 nearest points with d2<=4.0, stable (index order).
// ---------------------------------------------------------------------------
__global__ void ball_kernel(const float* __restrict__ pos, int n,
                            const float* __restrict__ qpos, int ns,
                            int* __restrict__ nbr) {
  int i = blockIdx.x * blockDim.x + threadIdx.x;
  if (i >= ns) return;
  float qx = qpos[i * 3 + 0], qy = qpos[i * 3 + 1], qz = qpos[i * 3 + 2];
  float bd[KNBR];
  int bi[KNBR];
  int cnt = 0;
  for (int j = 0; j < n; ++j) {
    float dx = pos[j * 3 + 0] - qx;
    float dy = pos[j * 3 + 1] - qy;
    float dz = pos[j * 3 + 2] - qz;
    float d2 = dx * dx + dy * dy + dz * dz;
    if (d2 <= 4.0f) {
      if (cnt < KNBR) {
        int p = cnt;
        while (p > 0 && bd[p - 1] > d2) { bd[p] = bd[p - 1]; bi[p] = bi[p - 1]; --p; }
        bd[p] = d2; bi[p] = j; ++cnt;
      } else if (d2 < bd[KNBR - 1]) {
        int p = KNBR - 1;
        while (p > 0 && bd[p - 1] > d2) { bd[p] = bd[p - 1]; bi[p] = bi[p - 1]; --p; }
        bd[p] = d2; bi[p] = j;
      }
    }
  }
  for (int k = 0; k < KNBR; ++k) nbr[(size_t)i * KNBR + k] = (k < cnt) ? bi[k] : -1;
}

__device__ __forceinline__ float safe_norm3(float x, float y, float z) {
  return sqrtf(fmaxf(x * x + y * y + z * z, 1e-12f));
}
__device__ __forceinline__ float angle3(float ax, float ay, float az,
                                        float bx, float by, float bz) {
  float cx = ay * bz - az * by;
  float cy = az * bx - ax * bz;
  float cz = ax * by - ay * bx;
  float dt = ax * bx + ay * by + az * bz;
  return atan2f(safe_norm3(cx, cy, cz), dt);
}

// Async memory->LDS staging (gfx1250): GLOBAL_LOAD_ASYNC_TO_LDS_B128 bypasses
// VGPRs, tracked by ASYNCcnt. Per ISA 10.2 the low 32 bits of a generic
// shared pointer are the workgroup-relative LDS byte address.
__device__ __forceinline__ void async_copy_to_lds(void* lds_dst, const void* gsrc,
                                                  int bytes, int tid, int nthreads) {
#if defined(__gfx1250__)
  uint32_t l = (uint32_t)(uintptr_t)lds_dst;
  const uint8_t* g = (const uint8_t*)gsrc;
  for (int c = tid; c < bytes / 16; c += nthreads) {
    uint32_t o = (uint32_t)c * 16u;
    asm volatile("global_load_async_to_lds_b128 %0, %1, off"
                 :
                 : "v"(l + o), "v"(g + o)
                 : "memory");
  }
#else
  uint32_t* d = (uint32_t*)lds_dst;
  const uint32_t* s = (const uint32_t*)gsrc;
  for (int c = tid; c < bytes / 4; c += nthreads) d[c] = s[c];
#endif
}
__device__ __forceinline__ void async_copy_wait() {
#if defined(__gfx1250__)
  asm volatile("s_wait_asynccnt 0x0" ::: "memory");
#endif
}

// ---------------------------------------------------------------------------
// Fused SA message kernel: gather x_j + PPF  ->  nn1 (68->68->68, relu x2 via
// WMMA, padded to 96) -> masked max over 32 neighbors -> agg f16 [ns,96].
// One wave per query; 8 queries per 256-thread block; weights async-staged in
// LDS. Invalid neighbors masked to 0 (exact: nn1 ends in relu, slot 0 valid).
// ---------------------------------------------------------------------------
__global__ __launch_bounds__(256)
void sa_msg_kernel(const float* __restrict__ x,     // [n,64]
                   const float* __restrict__ pos,   // [n,3]
                   const float* __restrict__ nrm,   // [n,3]
                   const float* __restrict__ qpos,  // [ns,3]
                   const float* __restrict__ qnrm,  // [ns,3]
                   const int* __restrict__ nbr,     // [ns,32]
                   const _Float16* __restrict__ w1, const float* __restrict__ b1,
                   const _Float16* __restrict__ w2, const float* __restrict__ b2,
                   _Float16* __restrict__ agg,      // [ns,96]
                   int ns) {
  __shared__ alignas(32) _Float16 sw1[DIMP * DIMP];
  __shared__ alignas(32) _Float16 sw2[DIMP * DIMP];
  __shared__ alignas(32) _Float16 stg[8][16 * DIMP];

  // async DMA of both weight matrices into LDS, then wait + barrier
  async_copy_to_lds(sw1, w1, DIMP * DIMP * 2, threadIdx.x, 256);
  async_copy_to_lds(sw2, w2, DIMP * DIMP * 2, threadIdx.x, 256);
  async_copy_wait();
  __syncthreads();

  int wave = threadIdx.x >> 5;
  int lane = threadIdx.x & 31;
  int lr = lane & 15;
  int hi = lane >> 4;
  int q = blockIdx.x * 8 + wave;
  if (q >= ns) return;

  int myn = nbr[(size_t)q * KNBR + lane];
  unsigned long long bal = __ballot(myn >= 0);
  int cnt = __popcll(bal & 0xffffffffull);

  float qpx = qpos[q * 3 + 0], qpy = qpos[q * 3 + 1], qpz = qpos[q * 3 + 2];
  float qnx = qnrm[q * 3 + 0], qny = qnrm[q * 3 + 1], qnz = qnrm[q * 3 + 2];

  float qmax[6] = {0.f, 0.f, 0.f, 0.f, 0.f, 0.f};
  _Float16* ss = &stg[wave][0];

  for (int t = 0; t < 2; ++t) {
    int m = t * 16 + lr;
    int j = nbr[(size_t)q * KNBR + m];
    bool valid = (j >= 0);
    float ppf0 = 0.f, ppf1 = 0.f, ppf2 = 0.f, ppf3 = 0.f;
    if (valid) {
      float dx = pos[j * 3 + 0] - qpx;
      float dy = pos[j * 3 + 1] - qpy;
      float dz = pos[j * 3 + 2] - qpz;
      float njx = nrm[j * 3 + 0], njy = nrm[j * 3 + 1], njz = nrm[j * 3 + 2];
      ppf0 = safe_norm3(dx, dy, dz);
      ppf1 = angle3(qnx, qny, qnz, dx, dy, dz);
      ppf2 = angle3(njx, njy, njz, dx, dy, dz);
      ppf3 = angle3(qnx, qny, qnz, njx, njy, njz);
    }
    // Build A fragments: row m, cols 0..63 = x[j], 64..67 = ppf, rest 0
    v16h af[3];
#pragma unroll
    for (int kc = 0; kc < 3; ++kc) {
#pragma unroll
      for (int h = 0; h < 16; ++h) {
        int col = kc * 32 + a_k_of_h(hi, h);
        float v = 0.0f;
        if (valid) {
          if (col < 64) v = x[(size_t)j * 64 + col];
          else if (col == 64) v = ppf0;
          else if (col == 65) v = ppf1;
          else if (col == 66) v = ppf2;
          else if (col == 67) v = ppf3;
        }
        af[kc][h] = (_Float16)v;
      }
    }
    // nn1 layer 0 -> LDS staging (bias + relu)
#pragma unroll
    for (int c = 0; c < 6; ++c) {
      v8f acc = {};
#pragma unroll
      for (int kc = 0; kc < 3; ++kc) {
        v16h b = *(const v16h*)&sw1[(c * 16 + lr) * DIMP + kc * 32 + hi * 16];
        acc = wmma_f16(af[kc], b, acc);
      }
      int n = c * 16 + lr;
      float bv = (n < 68) ? b1[n] : 0.0f;
#pragma unroll
      for (int r = 0; r < 8; ++r) {
        float v = fmaxf(acc[r] + bv, 0.0f);
        ss[(r + hi * 8) * DIMP + n] = (_Float16)v;
      }
    }
    // nn1 layer 1 (A from staging) + masked max-pool
    v16h a2[3];
#pragma unroll
    for (int kc = 0; kc < 3; ++kc) a2[kc] = load_a_frag(&ss[lr * DIMP + kc * 32], hi);
#pragma unroll
    for (int c = 0; c < 6; ++c) {
      v8f acc = {};
#pragma unroll
      for (int kc = 0; kc < 3; ++kc) {
        v16h b = *(const v16h*)&sw2[(c * 16 + lr) * DIMP + kc * 32 + hi * 16];
        acc = wmma_f16(a2[kc], b, acc);
      }
      int n = c * 16 + lr;
      float bv = (n < 68) ? b2[n] : 0.0f;
      float lm = 0.0f;
#pragma unroll
      for (int r = 0; r < 8; ++r) {
        float v = fmaxf(acc[r] + bv, 0.0f);
        int rowm = t * 16 + r + hi * 8;
        lm = fmaxf(lm, (rowm < cnt) ? v : 0.0f);
      }
      lm = fmaxf(lm, __shfl_xor(lm, 16, 32));
      qmax[c] = fmaxf(qmax[c], lm);
    }
  }
  if (hi == 0) {
#pragma unroll
    for (int c = 0; c < 6; ++c)
      agg[(size_t)q * DIMP + c * 16 + lr] = (_Float16)qmax[c];
  }
}

// ---------------------------------------------------------------------------
// FP: 3-NN inverse-distance interpolation + concat -> f16 [nf,128] GEMM input
// ---------------------------------------------------------------------------
__global__ void fp_build_kernel(const float* __restrict__ posF, int nf,
                                const float* __restrict__ posC, int nc,
                                const float* __restrict__ xc,     // [nc,64]
                                const float* __restrict__ xskip,  // [nf,64]
                                _Float16* __restrict__ outA) {    // [nf,128]
  int i = blockIdx.x * blockDim.x + threadIdx.x;
  if (i >= nf) return;
  float fx = posF[i * 3 + 0], fy = posF[i * 3 + 1], fz = posF[i * 3 + 2];
  float bd0 = 3.4e38f, bd1 = 3.4e38f, bd2 = 3.4e38f;
  int bi0 = 0, bi1 = 0, bi2 = 0;
  for (int j = 0; j < nc; ++j) {
    float dx = posC[j * 3 + 0] - fx;
    float dy = posC[j * 3 + 1] - fy;
    float dz = posC[j * 3 + 2] - fz;
    float d2 = dx * dx + dy * dy + dz * dz;
    if (d2 < bd0)      { bd2 = bd1; bi2 = bi1; bd1 = bd0; bi1 = bi0; bd0 = d2; bi0 = j; }
    else if (d2 < bd1) { bd2 = bd1; bi2 = bi1; bd1 = d2; bi1 = j; }
    else if (d2 < bd2) { bd2 = d2; bi2 = j; }
  }
  float w0 = 1.0f / fmaxf(bd0, 1e-16f);
  float w1 = 1.0f / fmaxf(bd1, 1e-16f);
  float w2 = 1.0f / fmaxf(bd2, 1e-16f);
  float wsum = w0 + w1 + w2;
  const float* c0 = xc + (size_t)bi0 * 64;
  const float* c1 = xc + (size_t)bi1 * 64;
  const float* c2 = xc + (size_t)bi2 * 64;
  for (int c = 0; c < 64; ++c) {
    float y = (c0[c] * w0 + c1[c] * w1 + c2[c] * w2) / wsum;
    outA[(size_t)i * 128 + c] = (_Float16)y;
  }
  for (int c = 0; c < 64; ++c)
    outA[(size_t)i * 128 + 64 + c] = (_Float16)xskip[(size_t)i * 64 + c];
}

// ---------------------------------------------------------------------------
// Host orchestration
// ---------------------------------------------------------------------------
extern "C" void kernel_launch(void* const* d_in, const int* in_sizes, int n_in,
                              void* d_out, int out_size, void* d_ws, size_t ws_size,
                              hipStream_t stream) {
  (void)in_sizes; (void)n_in; (void)out_size; (void)ws_size;
  const float* x_in = (const float*)d_in[0];
  const float* pos0 = (const float*)d_in[1];
  const float* nrm0 = (const float*)d_in[2];

  uint8_t* base = (uint8_t*)d_ws;
  size_t off = 0;
  auto alloc = [&](size_t bytes) -> void* {
    void* p = base + off;
    off += (bytes + 255) & ~(size_t)255;
    return p;
  };

  // --- weight prep: transpose + pad + f16 ---
  struct WDesc { int src, K, N, Kp, Np; };
  const WDesc wd[19] = {
      {3, 32, 64, 32, 64},   {5, 64, 64, 64, 64},                         // lin_in
      {7, 68, 68, 96, 96},   {9, 68, 68, 96, 96},  {11, 68, 64, 96, 64},  // sa0
      {13, 68, 68, 96, 96},  {15, 68, 68, 96, 96}, {17, 68, 64, 96, 64},  // sa1
      {19, 68, 68, 96, 96},  {21, 68, 68, 96, 96}, {23, 68, 64, 96, 64},  // sa2
      {25, 128, 128, 128, 128}, {27, 128, 64, 128, 64},                   // fp0
      {29, 128, 128, 128, 128}, {31, 128, 64, 128, 64},                   // fp1
      {33, 128, 128, 128, 128}, {35, 128, 64, 128, 64},                   // fp2
      {37, 64, 64, 64, 64},  {39, 64, 2, 64, 16}};                        // lin_out
  _Float16* wt[19];
  for (int i = 0; i < 19; ++i) {
    int tot = wd[i].Kp * wd[i].Np;
    wt[i] = (_Float16*)alloc((size_t)tot * sizeof(_Float16));
    prep_wt_kernel<<<(tot + 255) / 256, 256, 0, stream>>>(
        (const float*)d_in[wd[i].src], wt[i], wd[i].K, wd[i].N, wd[i].Kp, wd[i].Np);
  }

  const int nL[4] = {16384, 8192, 4096, 2048};
  float* h[4];
  for (int l = 0; l < 4; ++l) h[l] = (float*)alloc((size_t)nL[l] * NHF * sizeof(float));
  float* posA[3];
  float* nrmA[3];
  for (int l = 0; l < 3; ++l) {
    posA[l] = (float*)alloc((size_t)nL[l + 1] * 3 * sizeof(float));
    nrmA[l] = (float*)alloc((size_t)nL[l + 1] * 3 * sizeof(float));
  }
  const float* posL[4] = {pos0, posA[0], posA[1], posA[2]};
  const float* nrmL[4] = {nrm0, nrmA[0], nrmA[1], nrmA[2]};
  int* idx = (int*)alloc(8192 * sizeof(int));
  float* mind = (float*)alloc(16384 * sizeof(float));
  int* nbr = (int*)alloc((size_t)8192 * KNBR * sizeof(int));
  _Float16* agg = (_Float16*)alloc((size_t)8192 * DIMP * sizeof(_Float16));
  _Float16* fpA = (_Float16*)alloc((size_t)16384 * 128 * sizeof(_Float16));
  _Float16* fpB = (_Float16*)alloc((size_t)16384 * 128 * sizeof(_Float16));
  float* g[3];
  for (int i = 0; i < 3; ++i) g[i] = (float*)alloc((size_t)nL[i] * NHF * sizeof(float));

  // shape-dispatched WMMA GEMM (6 concrete (Kp,Np) shapes in this network)
  auto gemm = [&](const _Float16* A, const _Float16* WTp, const float* bias, int M,
                  int Kp, int Np, int Nreal, _Float16* oh, int npo, float* of, int relu) {
    int blocks = ((M / 16) + 7) / 8;
    if (Kp == 32 && Np == 64)
      gemm_t_kernel<1, 4><<<blocks, 256, 0, stream>>>(A, WTp, bias, M, Nreal, oh, npo, of, relu);
    else if (Kp == 64 && Np == 64)
      gemm_t_kernel<2, 4><<<blocks, 256, 0, stream>>>(A, WTp, bias, M, Nreal, oh, npo, of, relu);
    else if (Kp == 96 && Np == 64)
      gemm_t_kernel<3, 4><<<blocks, 256, 0, stream>>>(A, WTp, bias, M, Nreal, oh, npo, of, relu);
    else if (Kp == 128 && Np == 128)
      gemm_t_kernel<4, 8><<<blocks, 256, 0, stream>>>(A, WTp, bias, M, Nreal, oh, npo, of, relu);
    else if (Kp == 128 && Np == 64)
      gemm_t_kernel<4, 4><<<blocks, 256, 0, stream>>>(A, WTp, bias, M, Nreal, oh, npo, of, relu);
    else if (Kp == 64 && Np == 16)
      gemm_t_kernel<2, 1><<<blocks, 256, 0, stream>>>(A, WTp, bias, M, Nreal, oh, npo, of, relu);
  };

  // --- lin_in: x[16384,32] -> 64 relu -> 64 relu -> h0 ---
  pack_kernel<<<(16384 * 32 + 255) / 256, 256, 0, stream>>>(x_in, fpA, 16384, 32, 32);
  gemm(fpA, wt[0], (const float*)d_in[4], 16384, 32, 64, 64, fpB, 64, nullptr, 1);
  gemm(fpB, wt[1], (const float*)d_in[6], 16384, 64, 64, 64, nullptr, 0, h[0], 1);

  // --- SA levels ---
  for (int l = 0; l < 3; ++l) {
    int n = nL[l], ns = nL[l + 1];
    fps_kernel<<<1, FPS_T, 0, stream>>>(posL[l], n, ns, idx, mind);
    gather_kernel<<<(ns + 255) / 256, 256, 0, stream>>>(posL[l], nrmL[l], idx, ns,
                                                        (float*)posL[l + 1], (float*)nrmL[l + 1]);
    ball_kernel<<<(ns + 255) / 256, 256, 0, stream>>>(posL[l], n, posL[l + 1], ns, nbr);
    sa_msg_kernel<<<ns / 8, 256, 0, stream>>>(
        h[l], posL[l], nrmL[l], posL[l + 1], nrmL[l + 1], nbr,
        wt[2 + 3 * l], (const float*)d_in[8 + 6 * l],
        wt[3 + 3 * l], (const float*)d_in[10 + 6 * l], agg, ns);
    gemm(agg, wt[4 + 3 * l], (const float*)d_in[12 + 6 * l], ns, DIMP, 64, 64,
         nullptr, 0, h[l + 1], 1);
  }

  // --- FP levels (coarse -> fine) ---
  const float* cur = h[3];
  for (int i = 2; i >= 0; --i) {
    int nf = nL[i], nc = nL[i + 1];
    fp_build_kernel<<<(nf + 255) / 256, 256, 0, stream>>>(posL[i], nf, posL[i + 1], nc,
                                                          cur, h[i], fpA);
    gemm(fpA, wt[11 + 2 * i], (const float*)d_in[26 + 4 * i], nf, 128, 128, 128,
         fpB, 128, nullptr, 1);
    gemm(fpB, wt[12 + 2 * i], (const float*)d_in[28 + 4 * i], nf, 128, 64, 64,
         nullptr, 0, g[i], 1);
    cur = g[i];
  }

  // --- lin_out: 64 relu -> 2 (no act) ---
  pack_kernel<<<(16384 * 64 + 255) / 256, 256, 0, stream>>>(g[0], fpA, 16384, 64, 64);
  gemm(fpA, wt[17], (const float*)d_in[38], 16384, 64, 64, 64, fpB, 64, nullptr, 1);
  gemm(fpB, wt[18], (const float*)d_in[40], 16384, 64, 16, 2, nullptr, 0,
       (float*)d_out, 0);
}
